// TransMILDist_45947560132768
// MI455X (gfx1250) — matrix operand order repaired and checked by reading, not compiled, not run on hardware
//
#include <hip/hip_runtime.h>

// ---------------------------------------------------------------------------
// Model constants
// ---------------------------------------------------------------------------
#define BB      2
#define NPATCH  2048
#define INDIM   768
#define DIMC    512
#define DEPTHC  2
#define HEADSC  8
#define KNN_K   10
#define MLPD    2048
#define NCLS    2
#define DHC     64          // DIM / HEADS
#define NTOK    2049        // cls + patches
#define NTOKP   2080        // padded to multiple of 32
#define RTOT    (BB*NTOKP)  // 4160 total padded rows
#define NQT     (NTOKP/16)  // 130 query tiles
#define MASKW   (NTOKP/32)  // 65 mask words per row
#define LN_EPS  1e-5f

typedef unsigned short ushort_t;
typedef unsigned int   uint32;
typedef __attribute__((ext_vector_type(16))) __bf16 v16bf;
typedef __attribute__((ext_vector_type(8)))  float  v8f;

union BFrag { v16bf v; uint4 u4[2]; };

__device__ __forceinline__ ushort_t f2bf(float x) {
    unsigned u = __builtin_bit_cast(unsigned, x);
    unsigned r = (u + 0x7FFFu + ((u >> 16) & 1u)) >> 16;
    return (ushort_t)r;
}

__device__ __forceinline__ float gelu_exact(float x) {
    return 0.5f * x * (1.0f + erff(x * 0.70710678118654752f));
}

// ---------------------------------------------------------------------------
// fp32 -> bf16 elementwise convert
// ---------------------------------------------------------------------------
__global__ __launch_bounds__(256) void cvt_kernel(const float* __restrict__ src,
                                                  ushort_t* __restrict__ dst, int n) {
    int i = blockIdx.x * 256 + threadIdx.x;
    if (i < n) dst[i] = f2bf(src[i]);
}

// ---------------------------------------------------------------------------
// fp32 [K][N] -> bf16 [N][K] (weight transpose+convert)
// ---------------------------------------------------------------------------
__global__ __launch_bounds__(256) void wtrans_kernel(const float* __restrict__ src,
                                                     ushort_t* __restrict__ dst,
                                                     int K, int N) {
    __shared__ float tile[32][33];
    int kb = blockIdx.y * 32, nb = blockIdx.x * 32;
    int tx = threadIdx.x & 31, ty = threadIdx.x >> 5;   // ty: 0..7
    #pragma unroll
    for (int i = 0; i < 32; i += 8) {
        int k = kb + ty + i, n = nb + tx;
        tile[ty + i][tx] = (k < K && n < N) ? src[(size_t)k * N + n] : 0.0f;
    }
    __syncthreads();
    #pragma unroll
    for (int i = 0; i < 32; i += 8) {
        int n = nb + ty + i, k = kb + tx;
        if (n < N && k < K) dst[(size_t)n * K + k] = f2bf(tile[tx][ty + i]);
    }
}

// ---------------------------------------------------------------------------
// bf16 [R][C] -> bf16 [C][R]   (V transpose for attention B-fragments)
// ---------------------------------------------------------------------------
__global__ __launch_bounds__(256) void btrans_kernel(const ushort_t* __restrict__ src,
                                                     ushort_t* __restrict__ dst,
                                                     int Rr, int C) {
    __shared__ ushort_t tile[32][33];
    int rb = blockIdx.y * 32, cb = blockIdx.x * 32;
    int tx = threadIdx.x & 31, ty = threadIdx.x >> 5;
    #pragma unroll
    for (int i = 0; i < 32; i += 8) {
        int r = rb + ty + i, c = cb + tx;
        tile[ty + i][tx] = (r < Rr && c < C) ? src[(size_t)r * C + c] : (ushort_t)0;
    }
    __syncthreads();
    #pragma unroll
    for (int i = 0; i < 32; i += 8) {
        int c = cb + ty + i, r = rb + tx;
        if (c < C && r < Rr) dst[(size_t)c * Rr + r] = tile[tx][ty + i];
    }
}

// ---------------------------------------------------------------------------
// h init: cls token row + zero pad rows
// ---------------------------------------------------------------------------
__global__ __launch_bounds__(256) void init_h_kernel(const float* __restrict__ cls,
                                                     float* __restrict__ h) {
    int idx = blockIdx.x * 256 + threadIdx.x;        // B * 32 rows * 512
    if (idx >= BB * 32 * DIMC) return;
    int d  = idx & (DIMC - 1);
    int r  = idx >> 9;
    int b  = r >> 5;
    int qi = r & 31;
    int q  = (qi == 0) ? 0 : (NPATCH + qi);          // 0 or 2049..2079
    h[((size_t)b * NTOKP + q) * DIMC + d] = (qi == 0) ? cls[d] : 0.0f;
}

// ---------------------------------------------------------------------------
// allow-mask init: row 0 attends all valid; everyone attends col 0
// ---------------------------------------------------------------------------
__global__ __launch_bounds__(256) void mask_init_kernel(uint32* __restrict__ mask) {
    int idx = blockIdx.x * 256 + threadIdx.x;
    if (idx >= BB * NTOKP * MASKW) return;
    int w = idx % MASKW;
    int q = (idx / MASKW) % NTOKP;
    uint32 v = 0;
    if (q == 0) {
        int base = w * 32;
        if (base + 32 <= NTOK)      v = 0xFFFFFFFFu;
        else if (base < NTOK)       v = (1u << (NTOK - base)) - 1u;
    }
    if (w == 0) v |= 1u;    // cls column always allowed
    mask[idx] = v;
}

// ---------------------------------------------------------------------------
// KNN top-K neighbor mask (one block per (b, patch i))
// ---------------------------------------------------------------------------
__global__ __launch_bounds__(256) void knn_kernel(const float* __restrict__ coord,
                                                  uint32* __restrict__ mask) {
    int bi = blockIdx.x;
    int b = bi >> 11, i = bi & (NPATCH - 1);
    __shared__ float sd[NPATCH];
    __shared__ float rv[256];
    __shared__ int   ri[256];
    int t = threadIdx.x;
    float cx = coord[((size_t)b * NPATCH + i) * 2 + 0];
    float cy = coord[((size_t)b * NPATCH + i) * 2 + 1];
    for (int j = t; j < NPATCH; j += 256) {
        float dx = cx - coord[((size_t)b * NPATCH + j) * 2 + 0];
        float dy = cy - coord[((size_t)b * NPATCH + j) * 2 + 1];
        sd[j] = dx * dx + dy * dy;
    }
    __syncthreads();
    size_t mrow = ((size_t)b * NTOKP + 1 + i) * MASKW;
    for (int kk = 0; kk < KNN_K; ++kk) {
        float best = 1e30f; int bidx = 0x7FFFFFFF;
        for (int j = t; j < NPATCH; j += 256) {
            float v = sd[j];
            if (v < best || (v == best && j < bidx)) { best = v; bidx = j; }
        }
        rv[t] = best; ri[t] = bidx;
        __syncthreads();
        for (int s = 128; s > 0; s >>= 1) {
            if (t < s) {
                if (rv[t + s] < rv[t] || (rv[t + s] == rv[t] && ri[t + s] < ri[t])) {
                    rv[t] = rv[t + s]; ri[t] = ri[t + s];
                }
            }
            __syncthreads();
        }
        if (t == 0) {
            int j = ri[0];
            if (j < NPATCH) {
                sd[j] = 1e30f;
                int col = j + 1;
                mask[mrow + (col >> 5)] |= (1u << (col & 31));
            }
        }
        __syncthreads();
    }
}

// ---------------------------------------------------------------------------
// LayerNorm: fp32 rows -> bf16 rows (one block per row, DIM=512)
// ---------------------------------------------------------------------------
__global__ __launch_bounds__(256) void ln_kernel(const float* __restrict__ src,
                                                 const float* __restrict__ g,
                                                 const float* __restrict__ be,
                                                 ushort_t* __restrict__ dst) {
    int row = blockIdx.x, t = threadIdx.x;
    const float* x = src + (size_t)row * DIMC;
    float a0 = x[t], a1 = x[t + 256];
    __shared__ float red[256];
    red[t] = a0 + a1;
    __syncthreads();
    for (int s = 128; s > 0; s >>= 1) { if (t < s) red[t] += red[t + s]; __syncthreads(); }
    float mean = red[0] * (1.0f / DIMC);
    __syncthreads();
    float d0 = a0 - mean, d1 = a1 - mean;
    red[t] = d0 * d0 + d1 * d1;
    __syncthreads();
    for (int s = 128; s > 0; s >>= 1) { if (t < s) red[t] += red[t + s]; __syncthreads(); }
    float inv = rsqrtf(red[0] * (1.0f / DIMC) + LN_EPS);
    dst[(size_t)row * DIMC + t]       = f2bf(d0 * inv * g[t]       + be[t]);
    dst[(size_t)row * DIMC + t + 256] = f2bf(d1 * inv * g[t + 256] + be[t + 256]);
}

// ---------------------------------------------------------------------------
// Tiled WMMA GEMM, double-buffered LDS + register-staged global loads.
//   C[M,N] = act(A[M,K] x W^T + bias) (+residual)
//   A: bf16 row-major [M][K]; W: bf16 row-major TRANSPOSED [N][K]
//   Block tile 128x64, K-step 32, 8 waves: wave w -> 16-row slab, 4 N-tiles.
//   Pipeline per step: issue global loads for tile k+1 -> WMMA tile k from
//   LDS buf(cur) -> ds_store staged regs into buf(nxt) -> one barrier.
// ---------------------------------------------------------------------------
template<bool GELU, bool RESID, bool OUTF32>
__global__ __launch_bounds__(256) void gemm_kernel(const ushort_t* __restrict__ A,
                                                   const ushort_t* __restrict__ W,
                                                   const float* __restrict__ bias,
                                                   const float* __restrict__ resid,
                                                   float* __restrict__ outF,
                                                   ushort_t* __restrict__ outB,
                                                   int M, int N, int K) {
    __shared__ __align__(16) ushort_t sA[2][128][32];
    __shared__ __align__(16) ushort_t sB[2][64][32];
    int bm = blockIdx.y * 128, bn = blockIdx.x * 64;
    int t = threadIdx.x;
    int w = t >> 5, lane = t & 31, lm = lane & 15, lh = lane >> 4;

    // staging coordinates (per thread): 3x uint4 per 32-wide K tile
    const int ra0 = t >> 2;              // A rows 0..63
    const int ra1 = ra0 + 64;            // A rows 64..127
    const int ca  = (t & 3) * 8;
    const int rb  = t >> 2;              // W rows 0..63
    const bool a0ok = (bm + ra0) < M;
    const bool a1ok = (bm + ra1) < M;
    const bool bok  = (bn + rb)  < N;
    const ushort_t* aPtr0 = A + (size_t)(bm + ra0) * K + ca;
    const ushort_t* aPtr1 = A + (size_t)(bm + ra1) * K + ca;
    const ushort_t* wPtr  = W + (size_t)(bn + rb)  * K + ca;

    uint4 va0, va1, vb;
    const uint4 zero4 = {0u, 0u, 0u, 0u};

    // prologue: stage tile 0
    va0 = a0ok ? *(const uint4*)(aPtr0) : zero4;
    va1 = a1ok ? *(const uint4*)(aPtr1) : zero4;
    vb  = bok  ? *(const uint4*)(wPtr)  : zero4;
    *(uint4*)&sA[0][ra0][ca] = va0;
    *(uint4*)&sA[0][ra1][ca] = va1;
    *(uint4*)&sB[0][rb][ca]  = vb;
    __syncthreads();

    v8f acc[4] = {};
    for (int k0 = 0; k0 < K; k0 += 32) {
        const int cur = (k0 >> 5) & 1, nxt = cur ^ 1;
        const bool more = (k0 + 32) < K;
        // issue global loads for next tile (latency overlapped with WMMA)
        if (more) {
            va0 = a0ok ? *(const uint4*)(aPtr0 + k0 + 32) : zero4;
            va1 = a1ok ? *(const uint4*)(aPtr1 + k0 + 32) : zero4;
            vb  = bok  ? *(const uint4*)(wPtr  + k0 + 32) : zero4;
            if (k0 + 64 < K) {          // prefetch tile k+2 into cache
                if (a0ok) __builtin_prefetch(aPtr0 + k0 + 64, 0, 1);
                if (a1ok) __builtin_prefetch(aPtr1 + k0 + 64, 0, 1);
                if (bok)  __builtin_prefetch(wPtr  + k0 + 64, 0, 1);
            }
        }
        // compute from current buffer
        BFrag af;                                           // A frag 16x32
        af.u4[0] = *(const uint4*)&sA[cur][w * 16 + lm][lh * 8];
        af.u4[1] = *(const uint4*)&sA[cur][w * 16 + lm][lh * 8 + 16];
        #pragma unroll
        for (int tn = 0; tn < 4; ++tn) {                    // B frag 32x16
            BFrag bf;
            bf.u4[0] = *(const uint4*)&sB[cur][tn * 16 + lm][lh * 8];
            bf.u4[1] = *(const uint4*)&sB[cur][tn * 16 + lm][lh * 8 + 16];
            acc[tn] = __builtin_amdgcn_wmma_f32_16x16x32_bf16(
                false, af.v, false, bf.v, (short)0, acc[tn], false, false);
        }
        // store staged regs into the other buffer
        if (more) {
            *(uint4*)&sA[nxt][ra0][ca] = va0;
            *(uint4*)&sA[nxt][ra1][ca] = va1;
            *(uint4*)&sB[nxt][rb][ca]  = vb;
        }
        __syncthreads();
    }
    #pragma unroll
    for (int tn = 0; tn < 4; ++tn) {
        int col = bn + tn * 16 + lm;
        float bv = bias[col];
        #pragma unroll
        for (int r = 0; r < 8; ++r) {
            int row = bm + w * 16 + lh * 8 + r;
            if (row < M) {
                float v = acc[tn][r] + bv;
                if (GELU)  v = gelu_exact(v);
                if (RESID) v += resid[(size_t)row * N + col];
                if (OUTF32) outF[(size_t)row * N + col] = v;
                else        outB[(size_t)row * N + col] = f2bf(v);
            }
        }
    }
}

// ---------------------------------------------------------------------------
// Flash attention, one wave32 per (batch, head, 16-query tile)
//   q/k: bf16 [RTOT][DIM]; vt: bf16 [DIM][RTOT]; ob: bf16 [RTOT][DIM]
//   All K and V fragments for an iteration are hoisted into registers up
//   front so the global loads overlap the score WMMAs and the LDS softmax.
// ---------------------------------------------------------------------------
__global__ __launch_bounds__(32) void attn_kernel(const ushort_t* __restrict__ qb,
                                                  const ushort_t* __restrict__ kb,
                                                  const ushort_t* __restrict__ vt,
                                                  const uint32* __restrict__ maskw,
                                                  ushort_t* __restrict__ ob) {
    int qt = blockIdx.x % NQT;
    int hh = (blockIdx.x / NQT) & (HEADSC - 1);
    int b  = blockIdx.x / (NQT * HEADSC);
    int lane = threadIdx.x, lm = lane & 15, lh = lane >> 4;
    int rowbase = b * NTOKP + qt * 16;
    const float scale = 0.125f;                       // 1/sqrt(64)

    BFrag qa[2];                                      // Q frags, d-chunks of 32
    {
        const ushort_t* qp = qb + (size_t)(rowbase + lm) * DIMC + hh * DHC;
        #pragma unroll
        for (int c = 0; c < 2; ++c) {
            qa[c].u4[0] = *(const uint4*)(qp + c * 32 + lh * 8);
            qa[c].u4[1] = *(const uint4*)(qp + c * 32 + lh * 8 + 16);
        }
    }

    __shared__ float sM[16], sL[16], sAl[16];
    __shared__ float sS[16][32];
    __shared__ __align__(16) ushort_t sP[16][32];
    if (lane < 16) { sM[lane] = -1e30f; sL[lane] = 0.0f; }
    v8f oacc[4] = {};
    __syncthreads();

    for (int kt = 0; kt < MASKW; ++kt) {
        int k0 = kt * 32;
        // ---- hoisted loads: K frags (4), V frags (4), mask word ----
        BFrag kf[2][2];
        #pragma unroll
        for (int s = 0; s < 2; ++s) {
            const ushort_t* kp = kb + (size_t)(b * NTOKP + k0 + s * 16 + lm) * DIMC
                                    + hh * DHC + lh * 8;
            #pragma unroll
            for (int c = 0; c < 2; ++c) {
                kf[s][c].u4[0] = *(const uint4*)(kp + c * 32);
                kf[s][c].u4[1] = *(const uint4*)(kp + c * 32 + 16);
            }
        }
        BFrag vf[4];
        #pragma unroll
        for (int tn = 0; tn < 4; ++tn) {
            const ushort_t* vp = vt + (size_t)(hh * DHC + tn * 16 + lm) * RTOT
                                    + b * NTOKP + k0 + lh * 8;
            vf[tn].u4[0] = *(const uint4*)vp;
            vf[tn].u4[1] = *(const uint4*)(vp + 16);
        }
        uint32 mw = (lane < 16) ? maskw[(size_t)(rowbase + lane) * MASKW + kt] : 0u;

        // ---- scores: two 16x16 tiles ----
        #pragma unroll
        for (int s = 0; s < 2; ++s) {
            v8f sc = {};
            sc = __builtin_amdgcn_wmma_f32_16x16x32_bf16(
                false, qa[0].v, false, kf[s][0].v, (short)0, sc, false, false);
            sc = __builtin_amdgcn_wmma_f32_16x16x32_bf16(
                false, qa[1].v, false, kf[s][1].v, (short)0, sc, false, false);
            #pragma unroll
            for (int r = 0; r < 8; ++r) sS[r + lh * 8][s * 16 + lm] = sc[r];
        }
        __syncthreads();

        // ---- per-row online softmax (lanes 0..15, one row each) ----
        if (lane < 16) {
            int row = lane;
            float mprev = sM[row], tmax = -1e30f;
            #pragma unroll
            for (int j = 0; j < 32; ++j) {
                float v = ((mw >> j) & 1u) ? sS[row][j] * scale : -1e30f;
                sS[row][j] = v;
                tmax = fmaxf(tmax, v);
            }
            float mnew  = fmaxf(mprev, tmax);
            float alpha = __expf(mprev - mnew);
            float ssum = 0.0f;
            #pragma unroll
            for (int j = 0; j < 32; ++j) {
                float p = __expf(sS[row][j] - mnew);
                ssum += p;
                sP[row][j] = f2bf(p);
            }
            sM[row] = mnew; sL[row] = sL[row] * alpha + ssum; sAl[row] = alpha;
        }
        __syncthreads();

        // ---- rescale O, then O += P x V (V already in registers) ----
        float al[8];
        #pragma unroll
        for (int r = 0; r < 8; ++r) al[r] = sAl[r + lh * 8];
        #pragma unroll
        for (int tn = 0; tn < 4; ++tn)
            #pragma unroll
            for (int r = 0; r < 8; ++r) oacc[tn][r] *= al[r];
        BFrag pa;                                      // P as A-frag 16x32
        pa.u4[0] = *(const uint4*)&sP[lm][lh * 8];
        pa.u4[1] = *(const uint4*)&sP[lm][lh * 8 + 16];
        #pragma unroll
        for (int tn = 0; tn < 4; ++tn) {
            oacc[tn] = __builtin_amdgcn_wmma_f32_16x16x32_bf16(
                false, pa.v, false, vf[tn].v, (short)0, oacc[tn], false, false);
        }
        __syncthreads();
    }
    if (lane < 16) sL[lane] = 1.0f / sL[lane];
    __syncthreads();
    float il[8];
    #pragma unroll
    for (int r = 0; r < 8; ++r) il[r] = sL[r + lh * 8];
    #pragma unroll
    for (int tn = 0; tn < 4; ++tn)
        #pragma unroll
        for (int r = 0; r < 8; ++r)
            ob[(size_t)(rowbase + r + lh * 8) * DIMC + hh * DHC + tn * 16 + lm] =
                f2bf(oacc[tn][r] * il[r]);
}

// ---------------------------------------------------------------------------
// Final LN(row 0) + 2-class head (one block per batch)
// ---------------------------------------------------------------------------
__global__ __launch_bounds__(256) void head_kernel(const float* __restrict__ h,
                                                   const float* __restrict__ g,
                                                   const float* __restrict__ be,
                                                   const float* __restrict__ hw,
                                                   const float* __restrict__ hb,
                                                   float* __restrict__ out) {
    int b = blockIdx.x, t = threadIdx.x;
    const float* x = h + (size_t)b * NTOKP * DIMC;   // row 0 of batch b
    float a0 = x[t], a1 = x[t + 256];
    __shared__ float red[256];
    red[t] = a0 + a1;
    __syncthreads();
    for (int s = 128; s > 0; s >>= 1) { if (t < s) red[t] += red[t + s]; __syncthreads(); }
    float mean = red[0] * (1.0f / DIMC);
    __syncthreads();
    float d0 = a0 - mean, d1 = a1 - mean;
    red[t] = d0 * d0 + d1 * d1;
    __syncthreads();
    for (int s = 128; s > 0; s >>= 1) { if (t < s) red[t] += red[t + s]; __syncthreads(); }
    float inv = rsqrtf(red[0] * (1.0f / DIMC) + LN_EPS);
    __syncthreads();
    float h0 = d0 * inv * g[t] + be[t];
    float h1 = d1 * inv * g[t + 256] + be[t + 256];
    for (int c = 0; c < NCLS; ++c) {
        red[t] = h0 * hw[t * NCLS + c] + h1 * hw[(t + 256) * NCLS + c];
        __syncthreads();
        for (int s = 128; s > 0; s >>= 1) { if (t < s) red[t] += red[t + s]; __syncthreads(); }
        if (t == 0) out[b * NCLS + c] = red[0] + hb[c];
        __syncthreads();
    }
}

// ---------------------------------------------------------------------------
// Host orchestration
// ---------------------------------------------------------------------------
extern "C" void kernel_launch(void* const* d_in, const int* in_sizes, int n_in,
                              void* d_out, int out_size, void* d_ws, size_t ws_size,
                              hipStream_t stream) {
    (void)in_sizes; (void)n_in; (void)out_size; (void)ws_size;
    const float* x      = (const float*)d_in[0];
    const float* coord  = (const float*)d_in[1];
    const float* cls    = (const float*)d_in[3];
    const float* fc_w   = (const float*)d_in[4];
    const float* fc_b   = (const float*)d_in[5];
    const float* ln1_g  = (const float*)d_in[6];
    const float* ln1_b  = (const float*)d_in[7];
    const float* wq     = (const float*)d_in[8];
    const float* bq     = (const float*)d_in[9];
    const float* wk     = (const float*)d_in[10];
    const float* bk     = (const float*)d_in[11];
    const float* wv     = (const float*)d_in[12];
    const float* bv     = (const float*)d_in[13];
    const float* wo     = (const float*)d_in[14];
    const float* bo     = (const float*)d_in[15];
    const float* ln2_g  = (const float*)d_in[16];
    const float* ln2_b  = (const float*)d_in[17];
    const float* w1     = (const float*)d_in[18];
    const float* b1     = (const float*)d_in[19];
    const float* w2     = (const float*)d_in[20];
    const float* b2     = (const float*)d_in[21];
    const float* lnf_g  = (const float*)d_in[22];
    const float* lnf_b  = (const float*)d_in[23];
    const float* head_w = (const float*)d_in[24];
    const float* head_b = (const float*)d_in[25];
    float* out = (float*)d_out;

    // scratch carve-out
    char* p = (char*)d_ws;
    auto carve = [&](size_t bytes) -> void* {
        void* r = (void*)p;
        p += (bytes + 255) & ~(size_t)255;
        return r;
    };
    float*    hbuf = (float*)   carve((size_t)RTOT * DIMC * 4);
    ushort_t* xbf  = (ushort_t*)carve((size_t)BB * NPATCH * INDIM * 2);
    ushort_t* xn   = (ushort_t*)carve((size_t)RTOT * DIMC * 2);
    ushort_t* qb   = (ushort_t*)carve((size_t)RTOT * DIMC * 2);
    ushort_t* kbf  = (ushort_t*)carve((size_t)RTOT * DIMC * 2);
    ushort_t* vb   = (ushort_t*)carve((size_t)RTOT * DIMC * 2);
    ushort_t* vt   = (ushort_t*)carve((size_t)DIMC * RTOT * 2);
    ushort_t* obuf = (ushort_t*)carve((size_t)RTOT * DIMC * 2);
    ushort_t* ybf  = (ushort_t*)carve((size_t)RTOT * DIMC * 2);
    ushort_t* mlpb = (ushort_t*)carve((size_t)RTOT * MLPD * 2);
    uint32*   mask = (uint32*)  carve((size_t)BB * NTOKP * MASKW * 4);
    ushort_t* fcT  = (ushort_t*)carve((size_t)DIMC * INDIM * 2);
    ushort_t* wqT  = (ushort_t*)carve((size_t)DEPTHC * DIMC * DIMC * 2);
    ushort_t* wkT  = (ushort_t*)carve((size_t)DEPTHC * DIMC * DIMC * 2);
    ushort_t* wvT  = (ushort_t*)carve((size_t)DEPTHC * DIMC * DIMC * 2);
    ushort_t* woT  = (ushort_t*)carve((size_t)DEPTHC * DIMC * DIMC * 2);
    ushort_t* w1T  = (ushort_t*)carve((size_t)DEPTHC * MLPD * DIMC * 2);
    ushort_t* w2T  = (ushort_t*)carve((size_t)DEPTHC * DIMC * MLPD * 2);

    // --- input convert + weight transposes ---
    {
        int n = BB * NPATCH * INDIM;
        cvt_kernel<<<(n + 255) / 256, 256, 0, stream>>>(x, xbf, n);
    }
    wtrans_kernel<<<dim3(DIMC / 32, INDIM / 32), 256, 0, stream>>>(fc_w, fcT, INDIM, DIMC);
    for (int l = 0; l < DEPTHC; ++l) {
        size_t o = (size_t)l * DIMC * DIMC;
        wtrans_kernel<<<dim3(DIMC / 32, DIMC / 32), 256, 0, stream>>>(wq + o, wqT + o, DIMC, DIMC);
        wtrans_kernel<<<dim3(DIMC / 32, DIMC / 32), 256, 0, stream>>>(wk + o, wkT + o, DIMC, DIMC);
        wtrans_kernel<<<dim3(DIMC / 32, DIMC / 32), 256, 0, stream>>>(wv + o, wvT + o, DIMC, DIMC);
        wtrans_kernel<<<dim3(DIMC / 32, DIMC / 32), 256, 0, stream>>>(wo + o, woT + o, DIMC, DIMC);
        size_t om = (size_t)l * DIMC * MLPD;
        wtrans_kernel<<<dim3(MLPD / 32, DIMC / 32), 256, 0, stream>>>(w1 + om, w1T + om, DIMC, MLPD);
        wtrans_kernel<<<dim3(DIMC / 32, MLPD / 32), 256, 0, stream>>>(w2 + om, w2T + om, MLPD, DIMC);
    }

    // --- KNN allow mask ---
    {
        int nw = BB * NTOKP * MASKW;
        mask_init_kernel<<<(nw + 255) / 256, 256, 0, stream>>>(mask);
        knn_kernel<<<BB * NPATCH, 256, 0, stream>>>(coord, mask);
    }

    // --- h init (cls row, zero pad rows) + FC+GELU into h rows 1..2048 ---
    init_h_kernel<<<(BB * 32 * DIMC + 255) / 256, 256, 0, stream>>>(cls, hbuf);
    for (int b = 0; b < BB; ++b) {
        gemm_kernel<true, false, true><<<dim3(DIMC / 64, NPATCH / 128), 256, 0, stream>>>(
            xbf + (size_t)b * NPATCH * INDIM, fcT, fc_b, nullptr,
            hbuf + ((size_t)b * NTOKP + 1) * DIMC, nullptr, NPATCH, DIMC, INDIM);
    }

    dim3 gProj(DIMC / 64, (RTOT + 127) / 128);
    dim3 gMlp1(MLPD / 64, (RTOT + 127) / 128);

    for (int l = 0; l < DEPTHC; ++l) {
        size_t o  = (size_t)l * DIMC * DIMC;
        size_t om = (size_t)l * DIMC * MLPD;
        const float* bql = bq + l * DIMC;
        const float* bkl = bk + l * DIMC;
        const float* bvl = bv + l * DIMC;
        const float* bol = bo + l * DIMC;

        // LN1
        ln_kernel<<<RTOT, 256, 0, stream>>>(hbuf, ln1_g + l * DIMC, ln1_b + l * DIMC, xn);
        // Q, K, V projections (bf16 out)
        gemm_kernel<false, false, false><<<gProj, 256, 0, stream>>>(
            xn, wqT + o, bql, nullptr, nullptr, qb, RTOT, DIMC, DIMC);
        gemm_kernel<false, false, false><<<gProj, 256, 0, stream>>>(
            xn, wkT + o, bkl, nullptr, nullptr, kbf, RTOT, DIMC, DIMC);
        gemm_kernel<false, false, false><<<gProj, 256, 0, stream>>>(
            xn, wvT + o, bvl, nullptr, nullptr, vb, RTOT, DIMC, DIMC);
        // V transpose for attention B-frags
        btrans_kernel<<<dim3(DIMC / 32, RTOT / 32), 256, 0, stream>>>(vb, vt, RTOT, DIMC);
        // masked flash attention
        attn_kernel<<<BB * HEADSC * NQT, 32, 0, stream>>>(qb, kbf, vt, mask, obuf);
        // O projection + residual (h += o @ Wo + bo)
        gemm_kernel<false, true, true><<<gProj, 256, 0, stream>>>(
            obuf, woT + o, bol, hbuf, hbuf, nullptr, RTOT, DIMC, DIMC);
        // LN2
        ln_kernel<<<RTOT, 256, 0, stream>>>(hbuf, ln2_g + l * DIMC, ln2_b + l * DIMC, ybf);
        // MLP up + GELU
        gemm_kernel<true, false, false><<<gMlp1, 256, 0, stream>>>(
            ybf, w1T + om, b1 + l * MLPD, nullptr, nullptr, mlpb, RTOT, MLPD, DIMC);
        // MLP down + residual
        gemm_kernel<false, true, true><<<gProj, 256, 0, stream>>>(
            mlpb, w2T + om, b2 + l * DIMC, hbuf, hbuf, nullptr, RTOT, DIMC, MLPD);
    }

    // --- final LN + head ---
    head_kernel<<<BB, 256, 0, stream>>>(hbuf, lnf_g, lnf_b, head_w, head_b, out);
}
